// Head_34634616275588
// MI455X (gfx1250) — matrix operand order repaired and checked by reading, not compiled, not run on hardware
//
#include <hip/hip_runtime.h>
#include <stdint.h>

typedef __attribute__((ext_vector_type(16))) __bf16 v16bf;
typedef __attribute__((ext_vector_type(8)))  float  v8f;
typedef __attribute__((ext_vector_type(4)))  unsigned int u32x4;
typedef __attribute__((ext_vector_type(8)))  int i32x8;
typedef __attribute__((ext_vector_type(4)))  int i32x4;

#define T_SEQ   4096
#define DMODEL  1024
#define DHEAD   64
#define NBATCH  4

union V16 {
  uint32_t u[8];
  uint4    q[2];
  v16bf    v;
};

__device__ __forceinline__ uint16_t f2bf(float f) {
  uint32_t u = __float_as_uint(f);
  u += 0x7fffu + ((u >> 16) & 1u);          // round-to-nearest-even
  return (uint16_t)(u >> 16);
}
__device__ __forceinline__ uint32_t pack2(float lo, float hi) {
  return (uint32_t)f2bf(lo) | ((uint32_t)f2bf(hi) << 16);
}
__device__ __forceinline__ v8f wmma_bf16(v16bf a, v16bf b, v8f c) {
  return __builtin_amdgcn_wmma_f32_16x16x32_bf16(
      false, a, false, b, (short)0, c, false, false);
}

// TDM: load a [192 rows x 32 cols] bf16 tile (row stride 1024 elems) into LDS.
// D# group0: count=1, lds_addr, 57b global tile addr, type=2.
// D# group1: data_size=1(2B), tensor_dim0=1024, tensor_dim1=192,
//            tile_dim0=32, tile_dim1=192, tensor_dim0_stride=1024.
__device__ __forceinline__ void tdm_load_wchunk(const uint16_t* gsrc,
                                                uint32_t lds_off) {
  uint64_t ga = (uint64_t)(uintptr_t)gsrc;
  u32x4 g0 = {1u, lds_off, (uint32_t)ga,
              (uint32_t)((ga >> 32) & 0x01FFFFFFu) | 0x80000000u};
  i32x8 g1 = {0x00010000, (int)(1024u << 16), (int)(192u << 16),
              (int)(32u << 16), 192, 1024, 0, 0};
  i32x4 gz4 = {0, 0, 0, 0};
  i32x8 gz8 = {0, 0, 0, 0, 0, 0, 0, 0};
  __builtin_amdgcn_tensor_load_to_lds(g0, g1, gz4, gz4, gz8, 0);
}

// ---------------- Stage 0: transpose+convert weights to bf16 wT[3][64][1024]
__global__ void wconv_kernel(const float* __restrict__ wq,
                             const float* __restrict__ wk,
                             const float* __restrict__ wv,
                             uint16_t* __restrict__ wT) {
  int idx = blockIdx.x * 256 + threadIdx.x;   // 3*64*1024 = 196608
  int m   = idx >> 16;
  int rem = idx & 0xffff;
  int d   = rem >> 10;
  int kk  = rem & 1023;
  const float* w = (m == 0) ? wq : (m == 1) ? wk : wv;
  wT[idx] = f2bf(w[kk * DHEAD + d]);
}

// ---------------- Stage 1: q = x@wq+bq, k = x@wk+bk, vT = (x@wv+bv)^T (bf16)
// Weight K-chunks staged to LDS by the Tensor Data Mover, double buffered.
__global__ void __launch_bounds__(256) qkv_kernel(
    const float* __restrict__ x, const uint16_t* __restrict__ wT,
    const float* __restrict__ bq, const float* __restrict__ bk,
    const float* __restrict__ bv,
    uint16_t* __restrict__ qo, uint16_t* __restrict__ ko,
    uint16_t* __restrict__ vTo) {
  __shared__ uint16_t wlds[2][192 * 32];      // 2 x 12KB weight chunk buffers
  const int lane = threadIdx.x & 31;
  const int wave = threadIdx.x >> 5;
  const int half = lane >> 4;
  const int l16  = lane & 15;
  const int row0 = (blockIdx.x * 8 + wave) * 16;

  v8f z = {};
  v8f acc[3][4];
  for (int m = 0; m < 3; ++m)
    for (int n = 0; n < 4; ++n) acc[m][n] = z;

  // Prologue: DMA weight chunk 0 into buffer 0 (one wave issues for the block)
  if (wave == 0) {
    tdm_load_wchunk(wT, (uint32_t)(uintptr_t)&wlds[0][0]);
    __builtin_amdgcn_s_wait_tensorcnt(0);
  }
  __syncthreads();

  const float* xrow = x + (size_t)(row0 + l16) * DMODEL;
  for (int kc = 0; kc < 32; ++kc) {
    const int buf = kc & 1;
    // Prefetch next weight chunk into the other buffer via TDM
    if (wave == 0 && (kc + 1) < 32)
      tdm_load_wchunk(wT + (kc + 1) * 32,
                      (uint32_t)(uintptr_t)&wlds[buf ^ 1][0]);

    // A tile (16x32 bf16) from x: lane=M row, K at {8h..8h+7, 8h+16..8h+23}
    const float4* xp0 = (const float4*)(xrow + kc * 32 + half * 8);
    const float4* xp1 = (const float4*)(xrow + kc * 32 + half * 8 + 16);
    float4 f0 = xp0[0], f1 = xp0[1];
    float4 f2 = xp1[0], f3 = xp1[1];
    V16 a;
    a.u[0] = pack2(f0.x, f0.y); a.u[1] = pack2(f0.z, f0.w);
    a.u[2] = pack2(f1.x, f1.y); a.u[3] = pack2(f1.z, f1.w);
    a.u[4] = pack2(f2.x, f2.y); a.u[5] = pack2(f2.z, f2.w);
    a.u[6] = pack2(f3.x, f3.y); a.u[7] = pack2(f3.z, f3.w);

    const uint16_t* wb = &wlds[buf][0];
    for (int m = 0; m < 3; ++m) {
      // Batch the 4 B-tile LDS reads, then burst 4 WMMAs
      V16 bt[4];
      for (int nt = 0; nt < 4; ++nt) {
        const uint16_t* r = wb + (size_t)(m * 64 + nt * 16 + l16) * 32 + half * 16;
        bt[nt].q[0] = *(const uint4*)(r - half * 8);
        bt[nt].q[1] = *(const uint4*)(r - half * 8 + 16);
      }
      for (int nt = 0; nt < 4; ++nt)
        acc[m][nt] = wmma_bf16(a.v, bt[nt].v, acc[m][nt]);
    }

    if (wave == 0) __builtin_amdgcn_s_wait_tensorcnt(0);
    __syncthreads();
  }

  const int b   = row0 / T_SEQ;
  const int tr0 = row0 % T_SEQ;
  const float* bias[3] = {bq, bk, bv};
  for (int m = 0; m < 3; ++m) {
    for (int nt = 0; nt < 4; ++nt) {
      const int n = nt * 16 + l16;
      const float bb = bias[m][n];
      for (int r = 0; r < 8; ++r) {
        uint16_t h = f2bf(acc[m][nt][r] + bb);   // D: lane=N, M=r+8h
        const int rr = r + 8 * half;
        if (m == 0)      qo[(size_t)(row0 + rr) * DHEAD + n] = h;
        else if (m == 1) ko[(size_t)(row0 + rr) * DHEAD + n] = h;
        else             vTo[((size_t)b * DHEAD + n) * T_SEQ + (tr0 + rr)] = h;
      }
    }
  }
}

// ---------------- Stage 2: causal flash attention, 16 queries per wave
__global__ void __launch_bounds__(256) attn_kernel(
    const uint16_t* __restrict__ q, const uint16_t* __restrict__ k,
    const uint16_t* __restrict__ vT, float* __restrict__ out) {
  __shared__ uint16_t plds[8 * 16 * 32];       // per-wave 16x32 prob tile
  const int lane = threadIdx.x & 31;
  const int wave = threadIdx.x >> 5;
  const int half = lane >> 4;
  const int l16  = lane & 15;
  const int row0 = (blockIdx.x * 8 + wave) * 16;
  const int b    = row0 / T_SEQ;
  const int t0   = row0 % T_SEQ;

  const uint16_t* qb = q  + (size_t)row0 * DHEAD;
  const uint16_t* kb = k  + (size_t)b * T_SEQ * DHEAD;
  const uint16_t* vb = vT + (size_t)b * DHEAD * T_SEQ;
  uint16_t* pl = plds + wave * 16 * 32;

  // Preload Q as B operand of S' = K*Q^T: lane=N(t), K(d)=dc*32+16h+j contiguous
  V16 qB[2];
  for (int dc = 0; dc < 2; ++dc) {
    const uint4* p = (const uint4*)(qb + (size_t)l16 * DHEAD + dc * 32 + half * 16);
    qB[dc].q[0] = p[0]; qB[dc].q[1] = p[1];
  }

  v8f z = {};
  v8f acc[4] = {z, z, z, z};                  // 16(t) x 64(d) output, f32
  float m_run = -3.0e38f, l_run = 0.f;
  const int nch = (t0 + 16 + 31) >> 5;        // 32-key chunks (causal bound)

  for (int c = 0; c < nch; ++c) {
    const int s0 = c * 32;
    // Batch-load 4 K A-tiles (2 s-tiles x 2 d-chunks), then 4 WMMAs
    V16 ka[4];
    for (int st = 0; st < 2; ++st) {
      const uint16_t* krow = kb + (size_t)(s0 + st * 16 + l16) * DHEAD;
      for (int dc = 0; dc < 2; ++dc) {
        ka[st * 2 + dc].q[0] = *(const uint4*)(krow + dc * 32 + half * 16);
        ka[st * 2 + dc].q[1] = *(const uint4*)(krow + dc * 32 + half * 16 + 8);
      }
    }
    v8f sc[2];
    for (int st = 0; st < 2; ++st) {
      v8f s = z;
      s = wmma_bf16(ka[st * 2 + 0].v, qB[0].v, s);
      s = wmma_bf16(ka[st * 2 + 1].v, qB[1].v, s);
      sc[st] = s;
    }
    // scale + causal mask + online softmax (reduce over keys: in-lane + xor16)
    const int tg = t0 + l16;
    float cmax = -3.0e38f;
    for (int st = 0; st < 2; ++st)
      for (int r = 0; r < 8; ++r) {
        const int sg = s0 + st * 16 + r + 8 * half;
        float vv = sc[st][r] * 0.125f;        // 1/sqrt(64)
        if (sg > tg) vv = -1.0e9f;
        sc[st][r] = vv;
        cmax = fmaxf(cmax, vv);
      }
    cmax = fmaxf(cmax, __shfl_xor(cmax, 16, 32));
    const float m_new = fmaxf(m_run, cmax);
    const float alpha = __expf(m_run - m_new);
    float psum = 0.f;
    for (int st = 0; st < 2; ++st)
      for (int r = 0; r < 8; ++r) {
        const float p = __expf(sc[st][r] - m_new);
        psum += p;
        pl[l16 * 32 + st * 16 + r + 8 * half] = f2bf(p);  // LDS [t][s]
      }
    psum += __shfl_xor(psum, 16, 32);
    l_run = l_run * alpha + psum;
    m_run = m_new;
    // rescale accumulators (row scale broadcast matching D layout M=r+8h)
    for (int r = 0; r < 8; ++r) {
      const float ar = __shfl(alpha, r + half * 8, 32);
      for (int dt = 0; dt < 4; ++dt) acc[dt][r] *= ar;
    }
    // P as A operand from LDS (DS ops are in-order within a wave)
    V16 pa;
    pa.q[0] = *(const uint4*)(pl + l16 * 32 + half * 16);
    pa.q[1] = *(const uint4*)(pl + l16 * 32 + half * 16 + 8);
    // Batch-load 4 V B-tiles, then 4 WMMAs
    V16 vB[4];
    for (int dt = 0; dt < 4; ++dt) {
      const uint4* pv = (const uint4*)(vb + (size_t)(dt * 16 + l16) * T_SEQ
                                          + s0 + half * 16);
      vB[dt].q[0] = pv[0]; vB[dt].q[1] = pv[1];
    }
    for (int dt = 0; dt < 4; ++dt)
      acc[dt] = wmma_bf16(pa.v, vB[dt].v, acc[dt]);
  }

  for (int r = 0; r < 8; ++r) {
    const float lr  = __shfl(l_run, r + half * 8, 32);
    const float inv = 1.0f / lr;
    const int row   = row0 + r + 8 * half;
    for (int dt = 0; dt < 4; ++dt)
      out[(size_t)row * DHEAD + dt * 16 + l16] = acc[dt][r] * inv;
  }
}

extern "C" void kernel_launch(void* const* d_in, const int* in_sizes, int n_in,
                              void* d_out, int out_size, void* d_ws, size_t ws_size,
                              hipStream_t stream) {
  (void)in_sizes; (void)n_in; (void)out_size; (void)ws_size;
  const float* x  = (const float*)d_in[0];
  const float* wq = (const float*)d_in[1];
  const float* bq = (const float*)d_in[2];
  const float* wk = (const float*)d_in[3];
  const float* bk = (const float*)d_in[4];
  const float* wv = (const float*)d_in[5];
  const float* bv = (const float*)d_in[6];
  float* out = (float*)d_out;

  uint8_t* ws = (uint8_t*)d_ws;
  uint16_t* wT  = (uint16_t*)(ws);                          // 393,216 B
  uint16_t* qo  = (uint16_t*)(ws + 393216);                 // 2,097,152 B
  uint16_t* ko  = (uint16_t*)(ws + 393216 + 2097152);       // 2,097,152 B
  uint16_t* vTo = (uint16_t*)(ws + 393216 + 2 * 2097152);   // 2,097,152 B

  wconv_kernel<<<768, 256, 0, stream>>>(wq, wk, wv, wT);
  qkv_kernel<<<128, 256, 0, stream>>>(x, wT, bq, bk, bv, qo, ko, vTo);
  attn_kernel<<<128, 256, 0, stream>>>(qo, ko, vTo, out);
}
